// MultiheadGQA_27814208209300
// MI455X (gfx1250) — compile-verified, matrix operand order checked by reading, not run on hardware
//
#include <hip/hip_runtime.h>
#include <stdint.h>

// Problem constants (match reference)
#define B_    2
#define N_    2048
#define D_    2048
#define H_    16
#define KVH_  4
#define HD_   128
#define G_    4
#define KVD_  (KVH_ * HD_)   // 512

typedef __attribute__((ext_vector_type(16))) __bf16 v16bf;
typedef __attribute__((ext_vector_type(8)))  float  v8f;

union FragAB { v16bf v; uint32_t u[8]; };
union FragC  { v8f   v; float    f[8]; };

// dword index inside a 32-wide K chunk for 16-bit A/B fragments (ISA 7.12.2):
// lanes 0-15: dwords {0..3, 8..11}; lanes 16-31: {4..7, 12..15}
__device__ __forceinline__ int frag_dw(int i, int half) {
    return (i < 4) ? (i + 4 * half) : (i + 4 + 4 * half);
}

__device__ __forceinline__ uint16_t f2bf(float f) {
    uint32_t u = __builtin_bit_cast(uint32_t, f);
    u = (u + 0x7FFFu + ((u >> 16) & 1u)) >> 16;   // round-to-nearest-even
    return (uint16_t)u;
}

__device__ __forceinline__ v8f wmma_bf16(const FragAB& a, const FragAB& b, v8f c) {
    return __builtin_amdgcn_wmma_f32_16x16x32_bf16(
        false, a.v, false, b.v, (short)0, c, false, false);
}

// ---------------------------------------------------------------- conversions
__global__ __launch_bounds__(256) void cvt_f32_bf16(const float* __restrict__ in,
                                                    uint16_t* __restrict__ out, int n) {
    int i = blockIdx.x * blockDim.x + threadIdx.x;
    if (i < n) out[i] = f2bf(in[i]);
}

// Wt[n*K + k] = bf16(W[k*Nc + n])   (W is (K x Nc) row-major, in->out weights)
__global__ __launch_bounds__(256) void transpose_cvt(const float* __restrict__ W,
                                                     uint16_t* __restrict__ Wt,
                                                     int K, int Nc) {
    int i = blockIdx.x * blockDim.x + threadIdx.x;
    if (i >= Nc * K) return;
    int n = i / K, k = i - n * K;
    Wt[(size_t)n * K + k] = f2bf(W[(size_t)k * Nc + n]);
}

// ---------------------------------------------------------------- WMMA GEMM
// C = A(MxK,row) * Bt^T where Bt is (N x K) row-major. One wave -> 16x64 strip.
// Register ping-pong double buffering: next chunk's loads issue before current
// chunk's WMMAs so s_wait_loadcnt overlaps matrix work.
__global__ __launch_bounds__(128) void gemm_wmma_bf16(
    const uint16_t* __restrict__ A, const uint16_t* __restrict__ Bt,
    uint16_t* __restrict__ Cb, float* __restrict__ Cf,
    int M, int Nn, int K, float scale)
{
    const int lane = threadIdx.x & 31;
    const int wave = threadIdx.x >> 5;
    const int half = lane >> 4;
    const int l16  = lane & 15;
    const int m0 = blockIdx.y * 64 + wave * 16;
    const int n0 = blockIdx.x * 64;

    FragC c[4];
#pragma unroll
    for (int t = 0; t < 4; ++t)
#pragma unroll
        for (int r = 0; r < 8; ++r) c[t].f[r] = 0.f;

    const uint32_t* arow = (const uint32_t*)(A + (size_t)(m0 + l16) * K);
    const uint32_t* brow[4];
#pragma unroll
    for (int t = 0; t < 4; ++t)
        brow[t] = (const uint32_t*)(Bt + (size_t)(n0 + 16 * t + l16) * K);

    FragAB aP, bP[4], aQ, bQ[4];
    auto loadA = [&](FragAB& a, int kd) {
#pragma unroll
        for (int i = 0; i < 8; ++i) a.u[i] = arow[kd + frag_dw(i, half)];
    };
    auto loadB = [&](FragAB* bb, int kd) {
#pragma unroll
        for (int t = 0; t < 4; ++t)
#pragma unroll
            for (int i = 0; i < 8; ++i) bb[t].u[i] = brow[t][kd + frag_dw(i, half)];
    };

    loadA(aP, 0);
    loadB(bP, 0);
    for (int k0 = 0; k0 + 64 <= K; k0 += 64) {
        // prefetch A stream ~4KB ahead (L2 warm; speculative, drop on fault)
        __builtin_prefetch(arow + (k0 >> 1) + 1024, 0, 1);
        loadA(aQ, (k0 + 32) >> 1);
        loadB(bQ, (k0 + 32) >> 1);
#pragma unroll
        for (int t = 0; t < 4; ++t) c[t].v = wmma_bf16(aP, bP[t], c[t].v);
        if (k0 + 64 < K) {
            loadA(aP, (k0 + 64) >> 1);
            loadB(bP, (k0 + 64) >> 1);
        }
#pragma unroll
        for (int t = 0; t < 4; ++t) c[t].v = wmma_bf16(aQ, bQ[t], c[t].v);
    }

#pragma unroll
    for (int t = 0; t < 4; ++t)
#pragma unroll
        for (int r = 0; r < 8; ++r) {
            int row = m0 + r + 8 * half;
            int col = n0 + 16 * t + l16;
            float v = c[t].f[r] * scale;
            if (Cb) Cb[(size_t)row * Nn + col] = f2bf(v);
            else    Cf[(size_t)row * Nn + col] = v;
        }
}

// ---------------------------------------------------------------- attention
// Q: (B*N) x D (col = h*HD + d), K/V: (B*N) x KVD (col = kvh*HD + d)
// Out Ab: (B*N) x D bf16.  Block = 4 waves = 64 query rows of one (b,h).
__global__ __launch_bounds__(128) void attn_fwd(
    const uint16_t* __restrict__ Qb, const uint16_t* __restrict__ Kb,
    const uint16_t* __restrict__ Vb, uint16_t* __restrict__ Ab)
{
    __shared__ uint16_t lds_k[32][HD_];        // K tile, row-major      (8 KB)
    __shared__ uint16_t lds_vt[HD_][32];       // V tile transposed      (8 KB)
    __shared__ uint16_t lds_p[4][16][32];      // per-wave P scratch     (4 KB)

    const int lane = threadIdx.x & 31;
    const int wave = threadIdx.x >> 5;
    const int half = lane >> 4;
    const int l16  = lane & 15;
    const int tid  = threadIdx.x;

    const int bh  = blockIdx.y;
    const int b   = bh / H_;
    const int h   = bh % H_;
    const int kvh = h / G_;
    const int q0  = blockIdx.x * 64 + wave * 16;

    // Q fragments for all 4 K-steps (head_dim 128 = 4 x 32)
    FragAB qf[4];
    {
        const uint32_t* qrow =
            (const uint32_t*)(Qb + ((size_t)(b * N_ + q0 + l16) * D_ + h * HD_));
#pragma unroll
        for (int ks = 0; ks < 4; ++ks)
#pragma unroll
            for (int i = 0; i < 8; ++i)
                qf[ks].u[i] = qrow[ks * 16 + frag_dw(i, half)];
    }

    FragC o[8];
#pragma unroll
    for (int g = 0; g < 8; ++g)
#pragma unroll
        for (int r = 0; r < 8; ++r) o[g].f[r] = 0.f;
    float mrow[8], lrow[8];
#pragma unroll
    for (int r = 0; r < 8; ++r) { mrow[r] = -1e30f; lrow[r] = 0.f; }

    const int nkt = 2 * blockIdx.x + 2;   // causal: key tiles of 32 up to block diag

    for (int kt = 0; kt < nkt; ++kt) {
        const int s0 = kt * 32;
        // ---- cooperative staging -------------------------------------------
        {
            int kr = tid >> 2, c0 = (tid & 3) * 32;
            size_t base = (size_t)(b * N_ + s0 + kr) * KVD_ + kvh * HD_ + c0;

            // K tile: async DMA straight into LDS (ASYNCcnt path); the same
            // inst_offset is applied to both the LDS and global addresses.
            {
                const uint16_t* kg = Kb + base;
                uint32_t kdst = (uint32_t)(uintptr_t)(&lds_k[kr][c0]);
                asm volatile(
                    "global_load_async_to_lds_b128 %0, %1, off\n\t"
                    "global_load_async_to_lds_b128 %0, %1, off offset:16\n\t"
                    "global_load_async_to_lds_b128 %0, %1, off offset:32\n\t"
                    "global_load_async_to_lds_b128 %0, %1, off offset:48"
                    :: "v"(kdst), "v"(kg) : "memory");
            }
            // V tile transposed (needs lane shuffle, so global->reg->LDS)
            {
                const uint4* vs = (const uint4*)(Vb + base);
                union { uint4 v[4]; uint16_t e[32]; } tmp;
#pragma unroll
                for (int j = 0; j < 4; ++j) tmp.v[j] = vs[j];
#pragma unroll
                for (int i = 0; i < 32; ++i) lds_vt[c0 + i][kr] = tmp.e[i];
            }
            // async LDS writes must land before the barrier releases readers
            asm volatile("s_wait_asynccnt 0" ::: "memory");
        }
        __syncthreads();

        if (s0 <= q0 + 15) {          // wave has unmasked work in this tile
            // S = Q * K^T  (two 16x16 score fragments: keys s0..+15, s0+16..+31)
            FragC sc0, sc1;
#pragma unroll
            for (int r = 0; r < 8; ++r) { sc0.f[r] = 0.f; sc1.f[r] = 0.f; }
#pragma unroll
            for (int ks = 0; ks < 4; ++ks) {
                FragAB b0, b1;
                const uint32_t* k0p = (const uint32_t*)&lds_k[l16][ks * 32];
                const uint32_t* k1p = (const uint32_t*)&lds_k[l16 + 16][ks * 32];
#pragma unroll
                for (int i = 0; i < 8; ++i) {
                    b0.u[i] = k0p[frag_dw(i, half)];
                    b1.u[i] = k1p[frag_dw(i, half)];
                }
                sc0.v = wmma_bf16(qf[ks], b0, sc0.v);
                sc1.v = wmma_bf16(qf[ks], b1, sc1.v);
            }
            // causal mask + online softmax (rows r+8*half live across 16-lane half)
#pragma unroll
            for (int r = 0; r < 8; ++r) {
                int rowg = q0 + r + 8 * half;
                int c0g  = s0 + l16;
                if (c0g > rowg)      sc0.f[r] = -1e30f;
                if (c0g + 16 > rowg) sc1.f[r] = -1e30f;
            }
#pragma unroll
            for (int r = 0; r < 8; ++r) {
                float mx = fmaxf(sc0.f[r], sc1.f[r]);
#pragma unroll
                for (int off = 1; off < 16; off <<= 1)
                    mx = fmaxf(mx, __shfl_xor(mx, off, 32));
                float mnew = fmaxf(mrow[r], mx);
                float corr = __expf(mrow[r] - mnew);
                float p0 = __expf(sc0.f[r] - mnew);
                float p1 = __expf(sc1.f[r] - mnew);
                float rs = p0 + p1;
#pragma unroll
                for (int off = 1; off < 16; off <<= 1)
                    rs += __shfl_xor(rs, off, 32);
                lrow[r] = lrow[r] * corr + rs;
                mrow[r] = mnew;
#pragma unroll
                for (int g = 0; g < 8; ++g) o[g].f[r] *= corr;
                lds_p[wave][r + 8 * half][l16]      = f2bf(p0);
                lds_p[wave][r + 8 * half][l16 + 16] = f2bf(p1);
            }
            // intra-wave LDS RAW: make P stores visible before fragment reload
            asm volatile("s_wait_dscnt 0" ::: "memory");

            FragAB ap;
            const uint32_t* pr = (const uint32_t*)&lds_p[wave][l16][0];
#pragma unroll
            for (int i = 0; i < 8; ++i) ap.u[i] = pr[frag_dw(i, half)];
#pragma unroll
            for (int g = 0; g < 8; ++g) {
                FragAB bv;
                const uint32_t* vr = (const uint32_t*)&lds_vt[g * 16 + l16][0];
#pragma unroll
                for (int i = 0; i < 8; ++i) bv.u[i] = vr[frag_dw(i, half)];
                o[g].v = wmma_bf16(ap, bv, o[g].v);
            }
        }
        __syncthreads();
    }

    // normalize + store (bf16, GEMM-ready layout)
#pragma unroll
    for (int r = 0; r < 8; ++r) {
        float inv = 1.f / lrow[r];
        size_t rowg = (size_t)(b * N_ + q0 + r + 8 * half);
#pragma unroll
        for (int g = 0; g < 8; ++g)
            Ab[rowg * D_ + h * HD_ + g * 16 + l16] = f2bf(o[g].f[r] * inv);
    }
}

// ---------------------------------------------------------------- launcher
extern "C" void kernel_launch(void* const* d_in, const int* in_sizes, int n_in,
                              void* d_out, int out_size, void* d_ws, size_t ws_size,
                              hipStream_t stream)
{
    const float* x  = (const float*)d_in[0];
    const float* Wq = (const float*)d_in[1];
    const float* Wk = (const float*)d_in[2];
    const float* Wv = (const float*)d_in[3];
    const float* Wo = (const float*)d_in[4];
    float* out = (float*)d_out;

    char* ws = (char*)d_ws;
    size_t off = 0;
    auto carve = [&](size_t bytes) -> char* {
        char* p = ws + off;
        off += (bytes + 255) & ~(size_t)255;
        return p;
    };
    uint16_t* xb  = (uint16_t*)carve((size_t)B_ * N_ * D_ * 2);
    uint16_t* WqT = (uint16_t*)carve((size_t)D_ * D_ * 2);
    uint16_t* WkT = (uint16_t*)carve((size_t)KVD_ * D_ * 2);
    uint16_t* WvT = (uint16_t*)carve((size_t)KVD_ * D_ * 2);
    uint16_t* WoT = (uint16_t*)carve((size_t)D_ * D_ * 2);
    uint16_t* Qb  = (uint16_t*)carve((size_t)B_ * N_ * D_ * 2);
    uint16_t* Kb  = (uint16_t*)carve((size_t)B_ * N_ * KVD_ * 2);
    uint16_t* Vb  = (uint16_t*)carve((size_t)B_ * N_ * KVD_ * 2);
    uint16_t* Ab  = (uint16_t*)carve((size_t)B_ * N_ * D_ * 2);

    const int M = B_ * N_;                 // 4096
    const int nx = M * D_;
    cvt_f32_bf16<<<(nx + 255) / 256, 256, 0, stream>>>(x, xb, nx);
    transpose_cvt<<<(D_ * D_ + 255) / 256, 256, 0, stream>>>(Wq, WqT, D_, D_);
    transpose_cvt<<<(KVD_ * D_ + 255) / 256, 256, 0, stream>>>(Wk, WkT, D_, KVD_);
    transpose_cvt<<<(KVD_ * D_ + 255) / 256, 256, 0, stream>>>(Wv, WvT, D_, KVD_);
    transpose_cvt<<<(D_ * D_ + 255) / 256, 256, 0, stream>>>(Wo, WoT, D_, D_);

    dim3 blk(128);
    const float qscale = 0.08838834764831845f;   // 1/sqrt(128)
    gemm_wmma_bf16<<<dim3(D_ / 64, M / 64), blk, 0, stream>>>(
        xb, WqT, Qb, nullptr, M, D_, D_, qscale);
    gemm_wmma_bf16<<<dim3(KVD_ / 64, M / 64), blk, 0, stream>>>(
        xb, WkT, Kb, nullptr, M, KVD_, D_, 1.0f);
    gemm_wmma_bf16<<<dim3(KVD_ / 64, M / 64), blk, 0, stream>>>(
        xb, WvT, Vb, nullptr, M, KVD_, D_, 1.0f);

    attn_fwd<<<dim3(N_ / 64, B_ * H_), blk, 0, stream>>>(Qb, Kb, Vb, Ab);

    gemm_wmma_bf16<<<dim3(D_ / 64, M / 64), blk, 0, stream>>>(
        Ab, WoT, nullptr, out, M, D_, D_, 1.0f);
}
